// GraphAttentionNetwork_52836687675525
// MI455X (gfx1250) — matrix operand order repaired and checked by reading, not compiled, not run on hardware
//
#include <hip/hip_runtime.h>
#include <hip/hip_bf16.h>
#include <hip/hip_fp16.h>

// GAT forward for MI455X (gfx1250, wave32, WMMA).
// B=4, N=512, F=H=64, NHEADS=4. Outputs: x [B,N,H] then adj [B,N,N] (f32).

typedef __attribute__((ext_vector_type(16))) _Float16 v16h;
typedef __attribute__((ext_vector_type(8)))  _Float16 v8h;
typedef __attribute__((ext_vector_type(8)))  float    v8f;

#define BB     4
#define NN     512
#define FF     64
#define HH     64
#define NHEADS 4
#define NEGBIG (-1e9f)

// ---------------------------------------------------------------------------
// K0: input projections. One block per (b,n) row, 64 threads (one per out col).
//   u = x @ ep_W1[:F], v = x @ ep_W1[F:], h1t[k] = (x @ gat1_W[k])^T as f16.
// ---------------------------------------------------------------------------
__global__ void input_proj_kernel(const float* __restrict__ x,     // [B,N,F]
                                  const float* __restrict__ epW1,  // [2F,H]
                                  const float* __restrict__ g1W,   // [NHEADS,F,H]
                                  float* __restrict__ u,           // [B,N,H]
                                  float* __restrict__ v,           // [B,N,H]
                                  _Float16* __restrict__ h1t)      // [B,NHEADS,H,N]
{
    __shared__ float xr[FF];
    int row = blockIdx.x;           // 0..B*N-1
    int b = row / NN, n = row % NN;
    int f = threadIdx.x;            // 0..63
    xr[f] = x[(size_t)row * FF + f];
    __syncthreads();
    float au = 0.f, av = 0.f;
    float ah[NHEADS] = {0.f, 0.f, 0.f, 0.f};
    for (int c = 0; c < FF; ++c) {
        float xv = xr[c];
        au = fmaf(xv, epW1[c * HH + f], au);
        av = fmaf(xv, epW1[(FF + c) * HH + f], av);
#pragma unroll
        for (int k = 0; k < NHEADS; ++k)
            ah[k] = fmaf(xv, g1W[((size_t)k * FF + c) * HH + f], ah[k]);
    }
    u[(size_t)row * HH + f] = au;
    v[(size_t)row * HH + f] = av;
#pragma unroll
    for (int k = 0; k < NHEADS; ++k)
        h1t[(((size_t)b * NHEADS + k) * HH + f) * NN + n] = (_Float16)ah[k];
}

// ---------------------------------------------------------------------------
// Convert gat2_W [256,H] f32 -> transposed f16 [H,256] for WMMA B-fragments.
// ---------------------------------------------------------------------------
__global__ void convert_w2_kernel(const float* __restrict__ W2,
                                  _Float16* __restrict__ W2t)
{
    int i = blockIdx.x * blockDim.x + threadIdx.x;
    if (i < 256 * HH) {
        int c = i / HH, f = i % HH;
        W2t[(size_t)f * 256 + c] = (_Float16)W2[i];
    }
}

// ---------------------------------------------------------------------------
// Attention logit dots: si = h . a[:H], sj = h . a[H:], from transposed f16 h.
//   ht: [B, nheads, H, N]; a: [nheads, 2H]; si/sj: [B*nheads, N]
// ---------------------------------------------------------------------------
__global__ void head_dots_kernel(const _Float16* __restrict__ ht,
                                 const float* __restrict__ a, int a_stride,
                                 float* __restrict__ si, float* __restrict__ sj,
                                 int nheads)
{
    int n = blockIdx.x * blockDim.x + threadIdx.x;
    int k = blockIdx.y;
    int b = blockIdx.z;
    const _Float16* hp = ht + (((size_t)b * nheads + k) * HH) * NN + n;
    const float* ak = a + (size_t)k * a_stride;
    float accI = 0.f, accJ = 0.f;
#pragma unroll 8
    for (int c = 0; c < HH; ++c) {
        float hv = (float)hp[(size_t)c * NN];
        accI = fmaf(hv, ak[c], accI);
        accJ = fmaf(hv, ak[HH + c], accJ);
    }
    size_t o = ((size_t)b * nheads + k) * NN + n;
    si[o] = accI;
    sj[o] = accJ;
}

// ---------------------------------------------------------------------------
// K1: fused all-pairs edge MLP -> adj (never materializes [B,N,N,H]).
// 256 threads = one 16x16 (i,j) tile; u/v rows staged in LDS.
// ---------------------------------------------------------------------------
__global__ void edge_adj_kernel(const float* __restrict__ u,
                                const float* __restrict__ v,
                                const float* __restrict__ b1,
                                const float* __restrict__ w2,   // ep_W2[:,0], len H
                                const float* __restrict__ b2p,  // scalar
                                float* __restrict__ adj)        // [B,N,N]
{
    __shared__ float ut[16][HH], vt[16][HH], bsh[HH], wsh[HH];
    int b  = blockIdx.z;
    int i0 = blockIdx.y * 16, j0 = blockIdx.x * 16;
    int t  = threadIdx.x;            // 0..255
    int ti = t >> 4, tj = t & 15;
    for (int idx = t; idx < 16 * HH; idx += 256) {
        int r = idx / HH, c = idx % HH;
        ut[r][c] = u[((size_t)b * NN + i0 + r) * HH + c];
        vt[r][c] = v[((size_t)b * NN + j0 + r) * HH + c];
    }
    if (t < HH) { bsh[t] = b1[t]; wsh[t] = w2[t]; }
    __syncthreads();
    float acc = b2p[0];
#pragma unroll 8
    for (int c = 0; c < HH; ++c) {
        float tv = ut[ti][c] + vt[tj][c] + bsh[c];
        tv = fmaxf(tv, 0.f);
        acc = fmaf(tv, wsh[c], acc);
    }
    float w = 1.f / (1.f + __expf(-acc));
    int i = i0 + ti, j = j0 + tj;
    adj[((size_t)b * NN + i) * NN + j] = (i == j) ? 1.f : w;
}

// ---------------------------------------------------------------------------
// K2: per-row masked-softmax stats (rowmax, 1/rowsum). One wave32 per row.
// ---------------------------------------------------------------------------
__global__ void softmax_stats_kernel(const float* __restrict__ si,
                                     const float* __restrict__ sj,
                                     const float* __restrict__ adj,
                                     float* __restrict__ rowmax,
                                     float* __restrict__ rowinv,
                                     int nheads)
{
    int lane = threadIdx.x & 31;
    int wave = threadIdx.x >> 5;
    int i = blockIdx.x * (blockDim.x >> 5) + wave;
    int k = blockIdx.y, b = blockIdx.z;
    size_t hb = (size_t)b * nheads + k;
    float s_i = si[hb * NN + i];
    const float* sjr = sj + hb * NN;
    const float* ar  = adj + ((size_t)b * NN + i) * NN;
    float m = -3.402823e38f;
    for (int j = lane; j < NN; j += 32) {
        float e = s_i + sjr[j];
        e = e > 0.f ? e : 0.2f * e;
        float av = ar[j];
        float em = e * av + (1.f - av) * NEGBIG;
        m = fmaxf(m, em);
    }
#pragma unroll
    for (int off = 16; off > 0; off >>= 1)
        m = fmaxf(m, __shfl_xor(m, off, 32));
    float s = 0.f;
    for (int j = lane; j < NN; j += 32) {
        float e = s_i + sjr[j];
        e = e > 0.f ? e : 0.2f * e;
        float av = ar[j];
        float em = e * av + (1.f - av) * NEGBIG;
        s += __expf(em - m);
    }
#pragma unroll
    for (int off = 16; off > 0; off >>= 1)
        s += __shfl_xor(s, off, 32);
    if (lane == 0) {
        rowmax[hb * NN + i] = m;
        rowinv[hb * NN + i] = 1.f / s;
    }
}

// ---------------------------------------------------------------------------
// K3: fused attn recompute + (attn @ h) via v_wmma_f32_16x16x32_f16.
// Each wave owns a 16x64 output strip. Software-pipelined: fragments for
// K-step k+1 (loads + exp VALU, co-executable with XDL) are prefetched while
// the four WMMAs for step k run. Prefetch index wraps (branch-free, EXEC=~0).
//   IS_FINAL=false: apply ELU, write f16 into x1h [B,N,NHEADS*H]
//   IS_FINAL=true : write f32 into out [B,N,H]
// ---------------------------------------------------------------------------
template <bool IS_FINAL>
__global__ void attn_av_kernel(const float* __restrict__ si,
                               const float* __restrict__ sj,
                               const float* __restrict__ adj,
                               const float* __restrict__ rowmax,
                               const float* __restrict__ rowinv,
                               const _Float16* __restrict__ ht,  // [B,nheads,H,N]
                               _Float16* __restrict__ x1h,       // [B,N,NHEADS*H]
                               float* __restrict__ outx,         // [B,N,H]
                               int nheads)
{
    int lane = threadIdx.x & 31;
    int wave = threadIdx.x >> 5;
    int i0 = (blockIdx.x * 4 + wave) * 16;
    int k = blockIdx.y, b = blockIdx.z;
    size_t hb = (size_t)b * nheads + k;
    int m  = lane & 15;
    int hi = lane >> 4;                 // half-wave selector
    int i  = i0 + m;
    float s_i = si[hb * NN + i];
    float rm  = rowmax[hb * NN + i];
    float ri  = rowinv[hb * NN + i];
    const float* sjr = sj + hb * NN;
    const float* ar  = adj + ((size_t)b * NN + i) * NN;
    const _Float16* hbase = ht + hb * (size_t)HH * NN;

    // A fragment (k = 16*hi + p, same order as B): probs attn[i, j0..j0+31].
    auto make_a = [&](int j0) {
        v16h a;
#pragma unroll
        for (int p = 0; p < 16; ++p) {
            int kk = p + (hi << 4);
            int j = j0 + kk;
            float e = s_i + sjr[j];
            e = e > 0.f ? e : 0.2f * e;
            float av = ar[j];
            float em = e * av + (1.f - av) * NEGBIG;
            a[p] = (_Float16)(__expf(em - rm) * ri);
        }
        return a;
    };
    // B fragments: h[j0..j0+31, nt*16..+15]; ht transposed -> one 32B load.
    auto load_b = [&](int j0, v16h bf[4]) {
#pragma unroll
        for (int nt = 0; nt < 4; ++nt) {
            int col = nt * 16 + m;
            bf[nt] = *(const v16h*)(hbase + (size_t)col * NN + j0 + hi * 16);
        }
    };

    v8f acc[4] = {v8f{}, v8f{}, v8f{}, v8f{}};
    v16h aCur = make_a(0);
    v16h bCur[4];
    load_b(0, bCur);
    for (int j0 = 0; j0 < NN; j0 += 32) {
        int jn = (j0 + 32) & (NN - 1);       // wrap: branch-free prefetch
        v16h bNxt[4];
        load_b(jn, bNxt);
        v16h aNxt = make_a(jn);
#pragma unroll
        for (int nt = 0; nt < 4; ++nt)
            acc[nt] = __builtin_amdgcn_wmma_f32_16x16x32_f16(
                false, aCur, false, bCur[nt], (short)0, acc[nt], false, false);
        aCur = aNxt;
#pragma unroll
        for (int nt = 0; nt < 4; ++nt) bCur[nt] = bNxt[nt];
    }
    // C/D layout: VGPR r -> row i0 + r + hi*8, col nt*16 + (lane&15)
#pragma unroll
    for (int nt = 0; nt < 4; ++nt) {
#pragma unroll
        for (int r = 0; r < 8; ++r) {
            int orow = i0 + r + hi * 8;
            int ocol = nt * 16 + m;
            float val = acc[nt][r];
            if (IS_FINAL) {
                outx[((size_t)b * NN + orow) * HH + ocol] = val;
            } else {
                float e = val > 0.f ? val : (__expf(val) - 1.f);   // ELU
                x1h[((size_t)b * NN + orow) * (NHEADS * HH) + k * HH + ocol] =
                    (_Float16)e;
            }
        }
    }
}

// ---------------------------------------------------------------------------
// K4: h2 = x1 @ gat2_W via WMMA. x1h row-major f16 [B*N,256], W2t f16 [64,256].
// Double-buffered K loop; stores h2 transposed f16 [B,H,N].
// ---------------------------------------------------------------------------
__global__ void gemm_x1_kernel(const _Float16* __restrict__ x1h,
                               const _Float16* __restrict__ W2t,
                               _Float16* __restrict__ h2t)
{
    int lane = threadIdx.x & 31;
    int wave = threadIdx.x >> 5;
    int m0 = (blockIdx.x * 4 + wave) * 16;
    int m  = lane & 15;
    int hi = lane >> 4;
    int row = m0 + m;

    auto load_a = [&](int k0) {
        const _Float16* ap = x1h + (size_t)row * (NHEADS * HH) + k0 + hi * 16;
        v8h lo = *(const v8h*)ap;          // K = k0+hi*16 .. +7
        v8h hi8 = *(const v8h*)(ap + 8);   // K = k0+hi*16+8 .. +15
        v16h a;
#pragma unroll
        for (int p = 0; p < 8; ++p) { a[p] = lo[p]; a[8 + p] = hi8[p]; }
        return a;
    };
    auto load_b = [&](int k0, v16h bf[4]) {
#pragma unroll
        for (int nt = 0; nt < 4; ++nt) {
            int col = nt * 16 + m;
            bf[nt] = *(const v16h*)(W2t + (size_t)col * (NHEADS * HH) + k0 + hi * 16);
        }
    };

    v8f acc[4] = {v8f{}, v8f{}, v8f{}, v8f{}};
    v16h aCur = load_a(0);
    v16h bCur[4];
    load_b(0, bCur);
    for (int k0 = 0; k0 < NHEADS * HH; k0 += 32) {
        int kn = (k0 + 32) & (NHEADS * HH - 1);
        v16h bNxt[4];
        load_b(kn, bNxt);
        v16h aNxt = load_a(kn);
#pragma unroll
        for (int nt = 0; nt < 4; ++nt)
            acc[nt] = __builtin_amdgcn_wmma_f32_16x16x32_f16(
                false, aCur, false, bCur[nt], (short)0, acc[nt], false, false);
        aCur = aNxt;
#pragma unroll
        for (int nt = 0; nt < 4; ++nt) bCur[nt] = bNxt[nt];
    }
#pragma unroll
    for (int nt = 0; nt < 4; ++nt) {
#pragma unroll
        for (int r = 0; r < 8; ++r) {
            int orow = m0 + r + hi * 8;          // global row in [0, B*N)
            int ob = orow / NN, on = orow % NN;  // 16-tiles never straddle batches
            int ocol = nt * 16 + m;
            h2t[((size_t)ob * HH + ocol) * NN + on] = (_Float16)acc[nt][r];
        }
    }
}

// ---------------------------------------------------------------------------
extern "C" void kernel_launch(void* const* d_in, const int* in_sizes, int n_in,
                              void* d_out, int out_size, void* d_ws, size_t ws_size,
                              hipStream_t stream) {
    (void)in_sizes; (void)n_in; (void)out_size; (void)ws_size;
    const float* x    = (const float*)d_in[0];   // [B,N,F]
    const float* g1W  = (const float*)d_in[1];   // [NHEADS,F,H]
    const float* g1a  = (const float*)d_in[2];   // [NHEADS,2H]
    const float* g2W  = (const float*)d_in[3];   // [NHEADS*H,H]
    const float* g2a  = (const float*)d_in[4];   // [2H]
    const float* epW1 = (const float*)d_in[5];   // [2F,H]
    const float* epb1 = (const float*)d_in[6];   // [H]
    const float* epW2 = (const float*)d_in[7];   // [H,1]
    const float* epb2 = (const float*)d_in[8];   // [1]

    float* out_x = (float*)d_out;                       // [B,N,H]
    float* adj   = out_x + (size_t)BB * NN * HH;        // [B,N,N]

    // Workspace carve (all buffers fully overwritten every call).
    char* w = (char*)d_ws;
    auto carve = [&](size_t bytes) -> char* {
        char* p = w;
        w += (bytes + 255) & ~(size_t)255;
        return p;
    };
    float*    u    = (float*)carve((size_t)BB * NN * HH * 4);
    float*    v    = (float*)carve((size_t)BB * NN * HH * 4);
    _Float16* h1t  = (_Float16*)carve((size_t)BB * NHEADS * HH * NN * 2);
    _Float16* x1h  = (_Float16*)carve((size_t)BB * NN * NHEADS * HH * 2);
    _Float16* h2t  = (_Float16*)carve((size_t)BB * HH * NN * 2);
    _Float16* W2t  = (_Float16*)carve((size_t)HH * NHEADS * HH * 2);
    float*    s1i  = (float*)carve((size_t)BB * NHEADS * NN * 4);
    float*    s1j  = (float*)carve((size_t)BB * NHEADS * NN * 4);
    float*    s2i  = (float*)carve((size_t)BB * NN * 4);
    float*    s2j  = (float*)carve((size_t)BB * NN * 4);
    float*    rm1  = (float*)carve((size_t)BB * NHEADS * NN * 4);
    float*    ri1  = (float*)carve((size_t)BB * NHEADS * NN * 4);
    float*    rm2  = (float*)carve((size_t)BB * NN * 4);
    float*    ri2  = (float*)carve((size_t)BB * NN * 4);

    // 1) Input projections (u, v, per-head h1 transposed f16).
    input_proj_kernel<<<dim3(BB * NN), dim3(FF), 0, stream>>>(
        x, epW1, g1W, u, v, h1t);

    // 2) gat2 weight -> transposed f16.
    convert_w2_kernel<<<dim3((256 * HH + 255) / 256), dim3(256), 0, stream>>>(
        g2W, W2t);

    // 3) Fused edge MLP -> adj (written straight into d_out).
    edge_adj_kernel<<<dim3(NN / 16, NN / 16, BB), dim3(256), 0, stream>>>(
        u, v, epb1, epW2, epb2, adj);

    // 4) gat1 attention logit dots.
    head_dots_kernel<<<dim3(NN / 256, NHEADS, BB), dim3(256), 0, stream>>>(
        h1t, g1a, 2 * HH, s1i, s1j, NHEADS);

    // 5) gat1 softmax stats (one wave per row).
    softmax_stats_kernel<<<dim3(NN / 8, NHEADS, BB), dim3(256), 0, stream>>>(
        s1i, s1j, adj, rm1, ri1, NHEADS);

    // 6) gat1: fused attn-recompute + WMMA (attn @ h) + ELU -> x1 f16.
    attn_av_kernel<false><<<dim3(NN / 64, NHEADS, BB), dim3(128), 0, stream>>>(
        s1i, s1j, adj, rm1, ri1, h1t, x1h, nullptr, NHEADS);

    // 7) h2 = x1 @ gat2_W via WMMA (stored transposed f16).
    gemm_x1_kernel<<<dim3((BB * NN) / 64), dim3(128), 0, stream>>>(
        x1h, W2t, h2t);

    // 8) gat2 logit dots + stats.
    head_dots_kernel<<<dim3(NN / 256, 1, BB), dim3(256), 0, stream>>>(
        h2t, g2a, 2 * HH, s2i, s2j, 1);
    softmax_stats_kernel<<<dim3(NN / 8, 1, BB), dim3(256), 0, stream>>>(
        s2i, s2j, adj, rm2, ri2, 1);

    // 9) gat2: fused attn-recompute + WMMA -> final x (f32 into d_out).
    attn_av_kernel<true><<<dim3(NN / 64, 1, BB), dim3(128), 0, stream>>>(
        s2i, s2j, adj, rm2, ri2, h2t, nullptr, out_x, 1);
}
